// MixOrderAttention_66468913873092
// MI455X (gfx1250) — matrix operand order repaired and checked by reading, not compiled, not run on hardware
//
#include <hip/hip_runtime.h>
#include <hip/hip_bf16.h>

typedef __attribute__((ext_vector_type(16))) _Float16 v16h;
typedef __attribute__((ext_vector_type(8)))  _Float16 v8h;
typedef __attribute__((ext_vector_type(8)))  float    v8f;

// Problem constants (fixed by the reference)
constexpr int N  = 64;   // tokens per window
constexpr int C  = 256;  // channels
constexpr int H  = 8;    // heads
constexpr int HD = 32;   // head dim

// Transposed-f16 weight workspace layout (halves)
constexpr int WSQ_OFF   = 0;              // WqT   [256][256]
constexpr int WSKV_OFF  = 256 * 256;      // WkvT  [512][256]
constexpr int WSP_OFF   = WSKV_OFF + 512 * 256;  // WprojT [256][256]
constexpr int WS_HALVES = WSP_OFF + 256 * 256;   // 262144 halves = 512 KB

// LDS strides (elements). All chosen so v8h (16B) accesses stay 16B-aligned.
constexpr int XS  = 264; // f16 row stride for x/q/k/out_all
constexpr int QFS = 260; // f32 row stride for raw q/k GEMM results
constexpr int RS  = 65;  // f32 row stride for raw attention (scalar access only)
constexpr int AS  = 72;  // f16 row stride for normalized attention
constexpr int US  = 72;  // f16 row stride for transposed v/u buffers

// LDS pool layout (bytes, all 16B-aligned). Regions are phase-aliased.
constexpr int OFF_XH   = 0;
constexpr int OFF_QF   = OFF_XH  + N * XS * 2;          // 33792
constexpr int OFF_KF   = OFF_QF  + N * QFS * 4;         // 100352
constexpr int OFF_KH   = OFF_KF  + N * QFS * 4;         // 166912
constexpr int OFF_VT   = OFF_KH  + N * XS * 2;          // 200704
constexpr int OFF_ATTN = OFF_VT  + C * US * 2;          // 237568
constexpr int POOL_BYTES = OFF_ATTN + H * N * AS * 2;   // 311296 (~304 KB)

__device__ inline v16h cat8(v8h lo, v8h hi) {
  return __builtin_shufflevector(lo, hi, 0,1,2,3,4,5,6,7,8,9,10,11,12,13,14,15);
}

// A operand (16x32 f16, M x K): row M = m0+(lane&15);
// slots 0..7 -> K = kb + half*8 + s ; slots 8..15 -> K = kb + 16 + half*8 + (s-8)
__device__ inline v16h loadA16(const _Float16* base, int stride, int m0, int kb, int lane) {
  const _Float16* p = base + (m0 + (lane & 15)) * stride + kb + ((lane >> 4) << 3);
  v8h lo = *(const v8h*)p;
  v8h hi = *(const v8h*)(p + 16);
  return cat8(lo, hi);
}

// B operand (32x16 f16, K x N) where memory holds src[N][K] (K contiguous):
// N = n0+(lane&15); slots 0..7 -> K = kb + half*8 + s ; slots 8..15 -> K = kb + 16 + half*8 + (s-8). Works for LDS or global src.
__device__ inline v16h loadBT16(const _Float16* base, int stride, int n0, int kb, int lane) {
  const _Float16* p = base + (n0 + (lane & 15)) * stride + kb + ((lane >> 4) << 3);
  v8h lo = *(const v8h*)p;
  v8h hi = *(const v8h*)(p + 16);
  return cat8(lo, hi);
}

// Fallback B operand gathered from a row-major f32 weight matrix W[K][ld].
__device__ inline v16h loadBW(const float* W, int ld, int kb, int n0, int lane) {
  int n  = n0 + (lane & 15);
  int k0 = kb + ((lane >> 4) << 3);
  v16h b;
#pragma unroll
  for (int s = 0; s < 16; ++s) b[s] = (_Float16)W[(k0 + s + ((s >> 3) << 3)) * ld + n];
  return b;
}

// A operand = (I + sgn*A) fragment from normalized-attention f16 storage.
// NEG=false -> Lhat = I + A ; NEG=true -> L = I - A (vector negate; f16 WMMA
// has no neg_a modifier, only CNeg).
template<bool NEG>
__device__ inline v16h loadA_diag(const _Float16* A, int stride, int m0, int kb,
                                  int lane) {
  int i     = m0 + (lane & 15);
  int half8 = (lane >> 4) << 3;
  const _Float16* p = A + i * stride + kb + half8;
  v8h lo = *(const v8h*)p;
  v8h hi = *(const v8h*)(p + 16);
  v16h a = cat8(lo, hi);
  if (NEG) a = -a;
#pragma unroll
  for (int s = 0; s < 16; ++s) {
    int k = kb + half8 + ((s < 8) ? s : (s + 8));
    if (k == i) a[s] += (_Float16)1.0f;
  }
  return a;
}

// One wave: y[4][2] = (I +/- A) * u, A 64x64 f16 LDS, u transposed uT[32][US].
template<bool NEG>
__device__ inline void polyMM(const _Float16* attnH, const _Float16* BT, int lane,
                              v8f y[4][2]) {
#pragma unroll
  for (int mt = 0; mt < 4; ++mt) {
#pragma unroll
    for (int dt = 0; dt < 2; ++dt) {
      v8f acc = {};
#pragma unroll
      for (int kc = 0; kc < N; kc += 32) {
        v16h a  = loadA_diag<NEG>(attnH, AS, mt * 16, kc, lane);
        v16h bb = loadBT16(BT, US, dt * 16, kc, lane);
        acc = __builtin_amdgcn_wmma_f32_16x16x32_f16(false, a, false, bb, (short)0,
                                                     acc, false, false);
      }
      y[mt][dt] = acc;
    }
  }
}

// Store D fragments transposed into dstT[d][US] as f16: dstT = scale*y (+ addScale*addT)
__device__ inline void storeT(_Float16* dstT, int lane, const v8f y[4][2],
                              float scale, const _Float16* addT, float addScale) {
#pragma unroll
  for (int mt = 0; mt < 4; ++mt) {
#pragma unroll
    for (int dt = 0; dt < 2; ++dt) {
      int d  = dt * 16 + (lane & 15);
      int n0 = mt * 16 + ((lane >> 4) << 3);
      _Float16* p = dstT + d * US + n0;
      v8h vv;
#pragma unroll
      for (int r = 0; r < 8; ++r) {
        float v = y[mt][dt][r] * scale;
        if (addT) v += addScale * (float)addT[d * US + n0 + r];
        vv[r] = (_Float16)v;
      }
      *(v8h*)p = vv;
    }
  }
}

// ---------- Prep kernel: W(f32, row-major) -> WT(f16, [n][k], K-contiguous) ------
__global__ __launch_bounds__(256)
void cvt_weights_kernel(const float* __restrict__ Wq, const float* __restrict__ Wkv,
                        const float* __restrict__ Wproj, _Float16* __restrict__ ws) {
  int idx = blockIdx.x * 256 + threadIdx.x;      // 0 .. 262143
  if (idx < 65536) {
    int n = idx >> 8, k = idx & 255;
    ws[WSQ_OFF + n * 256 + k] = (_Float16)Wq[k * 256 + n];
  } else if (idx < 196608) {
    int t = idx - 65536;
    int n = t >> 8, k = t & 255;
    ws[WSKV_OFF + n * 256 + k] = (_Float16)Wkv[k * 512 + n];
  } else {
    int t = idx - 196608;
    int n = t >> 8, k = t & 255;
    ws[WSP_OFF + n * 256 + k] = (_Float16)Wproj[k * 256 + n];
  }
}

struct QkvTile {
  const _Float16* wt;  // transposed f16 weights (WS path)
  const float*    wg;  // raw f32 weights (gather fallback)
  const float*    bias;
  int ld, colbase, which, mt;
};

template<bool USE_WS>
__global__ __launch_bounds__(256)
void moa_fused_kernel(const float* __restrict__ x,
                      const float* __restrict__ Wq,    const float* __restrict__ bq,
                      const float* __restrict__ Wkv,   const float* __restrict__ bkv,
                      const float* __restrict__ Wgate, const float* __restrict__ bgate,
                      const float* __restrict__ Wproj, const float* __restrict__ bproj,
                      const float* __restrict__ bias_table,
                      const int*   __restrict__ rel_index,
                      const _Float16* __restrict__ wsw,
                      float* __restrict__ out) {
  __shared__ alignas(16) char pool[POOL_BYTES];
  __shared__ float mean_sh[C];
  __shared__ float gate_sh[H][4];
  __shared__ float dn_sh[H][N];

  const int tid  = threadIdx.x;
  const int w    = tid >> 5;   // wave id == head id in per-head phases
  const int lane = tid & 31;
  const int b    = blockIdx.x;

  _Float16* xh    = (_Float16*)(pool + OFF_XH);    // [N][XS]  x f16, later qh
  float*    qf    = (float*)   (pool + OFF_QF);    // [N][QFS] raw q
  float*    kf    = (float*)   (pool + OFF_KF);    // [N][QFS] raw k
  _Float16* kh    = (_Float16*)(pool + OFF_KH);    // [N][XS]  softmaxed k f16
  _Float16* vT    = (_Float16*)(pool + OFF_VT);    // [C][US]  v transposed f16
  _Float16* attnF = (_Float16*)(pool + OFF_ATTN);  // [H][N][AS] normalized A f16

  const _Float16* wsQ  = wsw + WSQ_OFF;
  const _Float16* wsKV = wsw + WSKV_OFF;
  const _Float16* wsP  = wsw + WSP_OFF;

  // ---------- Phase 0: load x -> f16 LDS, column means, per-head gates ----------
  {
    const float* xb = x + (size_t)b * (N * C);
    float s = 0.f;
#pragma unroll 4
    for (int m = 0; m < N; ++m) {
      float v = xb[m * C + tid];
      s += v;
      xh[m * XS + tid] = (_Float16)v;
    }
    mean_sh[tid] = s * (1.0f / 64.0f);
  }
  __syncthreads();
  if (tid < 32) {
    int h = tid >> 2, j = tid & 3;
    float acc = bgate[j];
#pragma unroll
    for (int d = 0; d < HD; ++d) acc += mean_sh[h * HD + d] * Wgate[d * 4 + j];
    gate_sh[h][j] = 1.0f / (1.0f + __expf(-acc));
  }
  __syncthreads();

  // ---------- Phase 1: QKV GEMMs, two interleaved tiles per iteration ----------
  auto decode = [&](int t) {
    QkvTile ti;
    ti.which = t >> 6;               // 0=q, 1=k, 2=v
    int tt = t & 63;
    ti.mt = tt >> 4;
    int nt = tt & 15;
    if (ti.which == 0) { ti.wt = wsQ;  ti.wg = Wq;  ti.bias = bq;  ti.ld = C;
                         ti.colbase = nt * 16; }
    else               { ti.wt = wsKV; ti.wg = Wkv; ti.bias = bkv; ti.ld = 2 * C;
                         ti.colbase = (ti.which == 2 ? C : 0) + nt * 16; }
    return ti;
  };
  auto storeQKV = [&](const QkvTile& ti, const v8f& acc) {
    int   cc = ti.colbase + (lane & 15);
    float bv = ti.bias[cc];
    int   i0 = ti.mt * 16 + ((lane >> 4) << 3);
    if (ti.which == 0) {
#pragma unroll
      for (int r = 0; r < 8; ++r) qf[(i0 + r) * QFS + cc] = acc[r] + bv;
    } else if (ti.which == 1) {
#pragma unroll
      for (int r = 0; r < 8; ++r) kf[(i0 + r) * QFS + cc] = acc[r] + bv;
    } else {
      int cp = cc - C;  // 0..255 == h*32+d
      v8h vv;
#pragma unroll
      for (int r = 0; r < 8; ++r) vv[r] = (_Float16)(acc[r] + bv);
      *(v8h*)(vT + cp * US + i0) = vv;   // vT[d-major][token]
    }
  };
  for (int t0 = w; t0 < 192; t0 += 16) {
    QkvTile ta = decode(t0);
    QkvTile tb = decode(t0 + 8);
    v8f acc0 = {}, acc1 = {};
#pragma unroll
    for (int kc = 0; kc < C; kc += 32) {
      v16h aA = loadA16(xh, XS, ta.mt * 16, kc, lane);
      v16h aB = loadA16(xh, XS, tb.mt * 16, kc, lane);
      v16h b0 = USE_WS ? loadBT16(ta.wt, C, ta.colbase, kc, lane)
                       : loadBW(ta.wg, ta.ld, kc, ta.colbase, lane);
      v16h b1 = USE_WS ? loadBT16(tb.wt, C, tb.colbase, kc, lane)
                       : loadBW(tb.wg, tb.ld, kc, tb.colbase, lane);
      acc0 = __builtin_amdgcn_wmma_f32_16x16x32_f16(false, aA, false, b0, (short)0,
                                                    acc0, false, false);
      acc1 = __builtin_amdgcn_wmma_f32_16x16x32_f16(false, aB, false, b1, (short)0,
                                                    acc1, false, false);
    }
    storeQKV(ta, acc0);
    storeQKV(tb, acc1);
  }
  __syncthreads();

  // ---------- Phase 2: softmax over hd for q and k -> f16 ----------
  _Float16* qh = xh;  // alias: x no longer needed
  for (int p = tid; p < N * H; p += 256) {
    int m = p >> 3, h = p & 7;
    {
      const float* src = qf + m * QFS + h * HD;
      _Float16* dst = qh + m * XS + h * HD;
      float vb[HD]; float mx = -1e30f;
#pragma unroll
      for (int d = 0; d < HD; ++d) { vb[d] = src[d]; mx = fmaxf(mx, vb[d]); }
      float s = 0.f;
#pragma unroll
      for (int d = 0; d < HD; ++d) { vb[d] = __expf(vb[d] - mx); s += vb[d]; }
      float inv = 1.0f / s;
#pragma unroll
      for (int d = 0; d < HD; ++d) dst[d] = (_Float16)(vb[d] * inv);
    }
    {
      const float* src = kf + m * QFS + h * HD;
      _Float16* dst = kh + m * XS + h * HD;
      float vb[HD]; float mx = -1e30f;
#pragma unroll
      for (int d = 0; d < HD; ++d) { vb[d] = src[d]; mx = fmaxf(mx, vb[d]); }
      float s = 0.f;
#pragma unroll
      for (int d = 0; d < HD; ++d) { vb[d] = __expf(vb[d] - mx); s += vb[d]; }
      float inv = 1.0f / s;
#pragma unroll
      for (int d = 0; d < HD; ++d) dst[d] = (_Float16)(vb[d] * inv);
    }
  }
  __syncthreads();

  // ---------- Phase 3: per-head attention build (one wave per head) ----------
  float* raw = (float*)(pool + OFF_QF) + w * (N * RS);  // aliases dead qf/kf
  const _Float16* qhh = qh + w * HD;
  const _Float16* khh = kh + w * HD;
#pragma unroll
  for (int it = 0; it < 4; ++it) {
    v16h a = loadA16(qhh, XS, it * 16, 0, lane);
#pragma unroll
    for (int jt = 0; jt < 4; ++jt) {
      v16h bb = loadBT16(khh, XS, jt * 16, 0, lane);
      v8f acc = {};
      acc = __builtin_amdgcn_wmma_f32_16x16x32_f16(false, a, false, bb, (short)0,
                                                   acc, false, false);
      int j  = jt * 16 + (lane & 15);
      int i0 = it * 16 + ((lane >> 4) << 3);
#pragma unroll
      for (int r = 0; r < 8; ++r) {
        int i = i0 + r;
        float bias = bias_table[rel_index[i * N + j] * H + w];
        raw[i * RS + j] = acc[r] + fmaxf(bias, 0.0f);
      }
    }
  }
  // symmetrize + degree normalization (reads raw only; lockstep within wave)
#pragma unroll
  for (int rr = 0; rr < 2; ++rr) {
    int i = lane + rr * 32;
    float dsum = 0.f;
    for (int j = 0; j < N; ++j) dsum += 0.5f * (raw[i * RS + j] + raw[j * RS + i]);
    dn_sh[w][i] = (dsum > 0.f) ? rsqrtf(dsum) : dsum;
  }
  _Float16* attnH = attnF + w * (N * AS);
#pragma unroll
  for (int rr = 0; rr < 2; ++rr) {
    int i = lane + rr * 32;
    float di = dn_sh[w][i];
    for (int j = 0; j < N; ++j) {
      float s = 0.5f * (raw[i * RS + j] + raw[j * RS + i]) * di * dn_sh[w][j];
      attnH[i * AS + j] = (_Float16)s;
    }
  }
  __syncthreads();  // raw region about to be reused across waves

  // ---------- Phase 4: Bernstein polynomial via Horner (6 matmuls/head) ----------
  // out = c0*g3*u3 + L(c1*g0*u2 + L(c2*g1*u1 + L(c3*g2*v))),  u_{i+1} = Lhat*u_i
  v8f outAcc[4][2];
  {
    _Float16* U1 = (_Float16*)(pool + OFF_QF) + w * (3 * HD * US);
    _Float16* U2 = U1 + HD * US;
    _Float16* ST = U2 + HD * US;
    const _Float16* vTh = vT + (w * HD) * US;
    const float c0 = 0.125f, c1 = 0.375f, c2 = 0.375f, c3 = 0.125f;
    const float g0 = gate_sh[w][0], g1 = gate_sh[w][1],
                g2 = gate_sh[w][2], g3 = gate_sh[w][3];
    v8f y[4][2];
    polyMM<false>(attnH, vTh, lane, y);  storeT(U1, lane, y, 1.f, nullptr, 0.f);   // u1
    polyMM<false>(attnH, U1,  lane, y);  storeT(U2, lane, y, 1.f, nullptr, 0.f);   // u2
    polyMM<false>(attnH, U2,  lane, y);                                            // u3
#pragma unroll
    for (int mt = 0; mt < 4; ++mt)
#pragma unroll
      for (int dt = 0; dt < 2; ++dt) outAcc[mt][dt] = y[mt][dt] * (c0 * g3);
    polyMM<true>(attnH, vTh, lane, y);   storeT(ST, lane, y, c3 * g2, U1, c2 * g1); // s_b
    polyMM<true>(attnH, ST,  lane, y);   storeT(U1, lane, y, 1.f,     U2, c1 * g0); // s_c
    polyMM<true>(attnH, U1,  lane, y);                                              // L*s_c
#pragma unroll
    for (int mt = 0; mt < 4; ++mt)
#pragma unroll
      for (int dt = 0; dt < 2; ++dt) outAcc[mt][dt] += y[mt][dt];
  }
  __syncthreads();  // all waves done reading their attention slice

  // ---------- Phase 5: gather heads -> out_all f16 (aliases attention region) ----
  _Float16* outAll = (_Float16*)(pool + OFF_ATTN);  // [N][XS]
#pragma unroll
  for (int mt = 0; mt < 4; ++mt)
#pragma unroll
    for (int dt = 0; dt < 2; ++dt) {
      int cc = w * HD + dt * 16 + (lane & 15);
      int i0 = mt * 16 + ((lane >> 4) << 3);
#pragma unroll
      for (int r = 0; r < 8; ++r)
        outAll[(i0 + r) * XS + cc] = (_Float16)outAcc[mt][dt][r];
    }
  __syncthreads();

  // ---------- Phase 6: output projection, two interleaved tiles ----------
  float* ob = out + (size_t)b * (N * C);
  for (int t0 = w; t0 < 64; t0 += 16) {
    int t1 = t0 + 8;
    int mt0 = t0 >> 4, nt0 = t0 & 15;
    int mt1 = t1 >> 4, nt1 = t1 & 15;
    v8f acc0 = {}, acc1 = {};
#pragma unroll
    for (int kc = 0; kc < C; kc += 32) {
      v16h aA = loadA16(outAll, XS, mt0 * 16, kc, lane);
      v16h aB = loadA16(outAll, XS, mt1 * 16, kc, lane);
      v16h b0 = USE_WS ? loadBT16(wsP, C, nt0 * 16, kc, lane)
                       : loadBW(Wproj, C, kc, nt0 * 16, lane);
      v16h b1 = USE_WS ? loadBT16(wsP, C, nt1 * 16, kc, lane)
                       : loadBW(Wproj, C, kc, nt1 * 16, lane);
      acc0 = __builtin_amdgcn_wmma_f32_16x16x32_f16(false, aA, false, b0, (short)0,
                                                    acc0, false, false);
      acc1 = __builtin_amdgcn_wmma_f32_16x16x32_f16(false, aB, false, b1, (short)0,
                                                    acc1, false, false);
    }
    {
      int cc = nt0 * 16 + (lane & 15);
      float bv = bproj[cc];
      int i0 = mt0 * 16 + ((lane >> 4) << 3);
#pragma unroll
      for (int r = 0; r < 8; ++r) ob[(i0 + r) * C + cc] = acc0[r] + bv;
    }
    {
      int cc = nt1 * 16 + (lane & 15);
      float bv = bproj[cc];
      int i0 = mt1 * 16 + ((lane >> 4) << 3);
#pragma unroll
      for (int r = 0; r < 8; ++r) ob[(i0 + r) * C + cc] = acc1[r] + bv;
    }
  }
}

extern "C" void kernel_launch(void* const* d_in, const int* in_sizes, int n_in,
                              void* d_out, int out_size, void* d_ws, size_t ws_size,
                              hipStream_t stream) {
  const float* x          = (const float*)d_in[0];
  const float* Wq         = (const float*)d_in[1];
  const float* bq         = (const float*)d_in[2];
  const float* Wkv        = (const float*)d_in[3];
  const float* bkv        = (const float*)d_in[4];
  const float* Wgate      = (const float*)d_in[5];
  const float* bgate      = (const float*)d_in[6];
  const float* Wproj      = (const float*)d_in[7];
  const float* bproj      = (const float*)d_in[8];
  const float* bias_table = (const float*)d_in[9];
  const int*   rel_index  = (const int*)d_in[10];
  (void)n_in; (void)out_size;

  int B = in_sizes[0] / (N * C);  // 1024

  if (ws_size >= (size_t)WS_HALVES * sizeof(_Float16)) {
    _Float16* wsw = (_Float16*)d_ws;
    cvt_weights_kernel<<<WS_HALVES / 256, 256, 0, stream>>>(Wq, Wkv, Wproj, wsw);
    moa_fused_kernel<true><<<B, 256, 0, stream>>>(x, Wq, bq, Wkv, bkv, Wgate, bgate,
                                                  Wproj, bproj, bias_table, rel_index,
                                                  wsw, (float*)d_out);
  } else {
    moa_fused_kernel<false><<<B, 256, 0, stream>>>(x, Wq, bq, Wkv, bkv, Wgate, bgate,
                                                   Wproj, bproj, bias_table, rel_index,
                                                   nullptr, (float*)d_out);
  }
}
